// GIN_35665408425999
// MI455X (gfx1250) — matrix operand order repaired and checked by reading, not compiled, not run on hardware
//
#include <hip/hip_runtime.h>

#define NODES   50000
#define EDGES   800000
#define C       128
#define NGRAPH  512
#define NOUT    5
#define BN_EPS  1e-5f
#define MTILES  (NODES / 16)          // 3125, exact

typedef __attribute__((ext_vector_type(16))) __bf16 v16bf;
typedef __attribute__((ext_vector_type(2)))  __bf16 v2bf;
typedef __attribute__((ext_vector_type(8)))  float  v8f;

static __device__ __forceinline__ unsigned short f2bf(float f) {
    union { float f; unsigned u; } v; v.f = f;
    unsigned r = v.u + 0x7FFFu + ((v.u >> 16) & 1u);   // round-to-nearest-even
    return (unsigned short)(r >> 16);
}

// Pack two f32 -> dword of 2 bf16. Prefer the dedicated builtin; otherwise use
// native __bf16 casts so isel can pick v_cvt_pk_bf16_f32 / v_cvt_bf16_f32.
static __device__ __forceinline__ unsigned pack2(float lo, float hi) {
#if __has_builtin(__builtin_amdgcn_cvt_pk_bf16_f32)
    auto r = __builtin_amdgcn_cvt_pk_bf16_f32(lo, hi);
    static_assert(sizeof(r) == 4, "cvt_pk_bf16 must return 4 bytes");
    unsigned u;
    __builtin_memcpy(&u, &r, 4);
    return u;
#else
    v2bf t;
    t.x = (__bf16)lo;
    t.y = (__bf16)hi;
    unsigned u;
    __builtin_memcpy(&u, &t, 4);
    return u;
#endif
}

// ---------------------------------------------------------------------------
// Convert the four 128x128 MLP weight matrices to bf16 (once per launch).
__global__ __launch_bounds__(256) void cvt_w(const float* __restrict__ w0,
                                             const float* __restrict__ w1,
                                             const float* __restrict__ w2,
                                             const float* __restrict__ w3,
                                             unsigned short* __restrict__ out) {
    int i = blockIdx.x * 256 + threadIdx.x;            // 0 .. 65535
    int which = i >> 14, off = i & 16383;
    const float* s = (which == 0) ? w0 : (which == 1) ? w1 : (which == 2) ? w2 : w3;
    out[i] = f2bf(s[off]);
}

// ---------------------------------------------------------------------------
// One wave per edge: msg = relu(x[src] + ew*We + be), scatter-add into aggr.
__global__ __launch_bounds__(256) void msg_scatter(const float* __restrict__ x,
                                                   const int*   __restrict__ ei,
                                                   const float* __restrict__ ew,
                                                   const float* __restrict__ We,
                                                   const float* __restrict__ be,
                                                   float* __restrict__ aggr) {
    int wave = (blockIdx.x * 256 + threadIdx.x) >> 5;  // edge id (exact: 100000*8)
    int lane = threadIdx.x & 31;
    int s = ei[wave];
    int d = ei[EDGES + wave];
    float w = ew[wave];
    float4 xv = ((const float4*)(x + (size_t)s * C))[lane];
    float4 wv = ((const float4*)We)[lane];
    float4 bv = ((const float4*)be)[lane];
    float m0 = fmaxf(fmaf(w, wv.x, bv.x) + xv.x, 0.f);
    float m1 = fmaxf(fmaf(w, wv.y, bv.y) + xv.y, 0.f);
    float m2 = fmaxf(fmaf(w, wv.z, bv.z) + xv.z, 0.f);
    float m3 = fmaxf(fmaf(w, wv.w, bv.w) + xv.w, 0.f);
    float* ap = aggr + (size_t)d * C + lane * 4;
    atomicAdd(ap + 0, m0);
    atomicAdd(ap + 1, m1);
    atomicAdd(ap + 2, m2);
    atomicAdd(ap + 3, m3);
}

// ---------------------------------------------------------------------------
// Fused GEMM: out[m, o] = sum_k A[m,k] * W[o,k] + bias[o]
// One wave computes a full 16x128 output stripe: 8 f32 accumulators,
// A fragment built in-register from f32 sources (fusing h-build / BN-act),
// B fragments loaded straight from row-major bf16 weights in ISA layout.
//
// ASRC: 1 = A = bf16(a0 + a1)                    (GINE h = x + aggr)
//       2 = A = bf16(relu(a0*scale + shift))     (BatchNorm affine + ReLU)
// EPI:  0 = store f32 + BatchNorm sum/sumsq atomics
//       1 = relu + store f32
//       2 = relu + segment-max atomicMax into pooled (values >= 0)
template<int ASRC>
static __device__ __forceinline__ uint4 a_chunk(const float* a0, const float* a1,
                                                const float* scale, const float* shift,
                                                size_t rowoff, int k0) {
    float4 u0 = *(const float4*)(a0 + rowoff + k0);
    float4 u1 = *(const float4*)(a0 + rowoff + k0 + 4);
    float4 s0, s1;
    if (ASRC == 1) {
        float4 v0 = *(const float4*)(a1 + rowoff + k0);
        float4 v1 = *(const float4*)(a1 + rowoff + k0 + 4);
        s0 = make_float4(u0.x + v0.x, u0.y + v0.y, u0.z + v0.z, u0.w + v0.w);
        s1 = make_float4(u1.x + v1.x, u1.y + v1.y, u1.z + v1.z, u1.w + v1.w);
    } else {
        float4 c0 = *(const float4*)(scale + k0);
        float4 c1 = *(const float4*)(scale + k0 + 4);
        float4 h0 = *(const float4*)(shift + k0);
        float4 h1 = *(const float4*)(shift + k0 + 4);
        s0 = make_float4(fmaxf(fmaf(u0.x, c0.x, h0.x), 0.f),
                         fmaxf(fmaf(u0.y, c0.y, h0.y), 0.f),
                         fmaxf(fmaf(u0.z, c0.z, h0.z), 0.f),
                         fmaxf(fmaf(u0.w, c0.w, h0.w), 0.f));
        s1 = make_float4(fmaxf(fmaf(u1.x, c1.x, h1.x), 0.f),
                         fmaxf(fmaf(u1.y, c1.y, h1.y), 0.f),
                         fmaxf(fmaf(u1.z, c1.z, h1.z), 0.f),
                         fmaxf(fmaf(u1.w, c1.w, h1.w), 0.f));
    }
    uint4 q;
    q.x = pack2(s0.x, s0.y);
    q.y = pack2(s0.z, s0.w);
    q.z = pack2(s1.x, s1.y);
    q.w = pack2(s1.z, s1.w);
    return q;
}

template<int ASRC, int EPI>
__global__ __launch_bounds__(256) void gemm_fused(const float* a0, const float* a1,
                                                  const float* __restrict__ scale,
                                                  const float* __restrict__ shift,
                                                  const unsigned short* __restrict__ W,
                                                  const float* __restrict__ bias,
                                                  float* out,
                                                  float* __restrict__ bnsum,
                                                  float* __restrict__ bnss,
                                                  const int* __restrict__ batch,
                                                  float* __restrict__ pooled) {
    int gw = (blockIdx.x * 256 + threadIdx.x) >> 5;    // wave id = M tile
    if (gw >= MTILES) return;                          // whole-wave exit (EXEC all-1)
    int lane = threadIdx.x & 31;
    int half = lane >> 4;
    int l16  = lane & 15;
    int m_base = gw * 16;
    size_t rowoff = (size_t)(m_base + l16) * C;        // A row for this lane

    v8f acc[8];
#pragma unroll
    for (int n = 0; n < 8; n++) acc[n] = v8f{};

    union { v16bf v; uint4 q[2]; } a, b;
#pragma unroll
    for (int kb = 0; kb < C; kb += 32) {
        // A (16x32 bf16): lane -> row m_base+l16, chunks at kb+8*half, +16
        a.q[0] = a_chunk<ASRC>(a0, a1, scale, shift, rowoff, kb + 8 * half);
        a.q[1] = a_chunk<ASRC>(a0, a1, scale, shift, rowoff, kb + 8 * half + 16);
#pragma unroll
        for (int n = 0; n < 8; n++) {
            // B (32x16 bf16): lane -> col n*16+l16, chunks at kb+8*half, +16
            const unsigned short* brow = W + (size_t)(n * 16 + l16) * C + kb + 8 * half;
            b.q[0] = *(const uint4*)(brow);
            b.q[1] = *(const uint4*)(brow + 16);
            acc[n] = __builtin_amdgcn_wmma_f32_16x16x32_bf16(false, a.v, false, b.v,
                                                             (short)0, acc[n], false, false);
        }
    }

    int bj[8];
    if (EPI == 2) {
#pragma unroll
        for (int j = 0; j < 8; j++) bj[j] = batch[m_base + 8 * half + j];
    }

#pragma unroll
    for (int n = 0; n < 8; n++) {
        int col = n * 16 + l16;                        // D: n=lane%16, rows 8*half+j
        float bv = bias[col];
        if (EPI == 0) {
            float s = 0.f, ss = 0.f;
#pragma unroll
            for (int j = 0; j < 8; j++) {
                float v = acc[n][j] + bv;
                out[(size_t)(m_base + 8 * half + j) * C + col] = v;
                s += v;
                ss += v * v;
            }
            s  += __shfl_down(s, 16, 32);
            ss += __shfl_down(ss, 16, 32);
            if (half == 0) {
                atomicAdd(bnsum + col, s);
                atomicAdd(bnss + col, ss);
            }
        } else if (EPI == 1) {
#pragma unroll
            for (int j = 0; j < 8; j++) {
                float v = fmaxf(acc[n][j] + bv, 0.f);
                out[(size_t)(m_base + 8 * half + j) * C + col] = v;
            }
        } else {
#pragma unroll
            for (int j = 0; j < 8; j++) {
                float v = fmaxf(acc[n][j] + bv, 0.f);
                atomicMax((int*)(pooled + (size_t)bj[j] * C + col), __float_as_int(v));
            }
        }
    }
}

// ---------------------------------------------------------------------------
__global__ void bn_fin(const float* __restrict__ sum, const float* __restrict__ ss,
                       const float* __restrict__ g, const float* __restrict__ bt,
                       float* __restrict__ scale, float* __restrict__ shift) {
    int c = threadIdx.x;                   // 128 threads
    const float inv_n = 1.0f / (float)NODES;
    float mean = sum[c] * inv_n;
    float var  = ss[c] * inv_n - mean * mean;
    float sc   = g[c] * rsqrtf(var + BN_EPS);
    scale[c] = sc;
    shift[c] = bt[c] - mean * sc;
}

__global__ __launch_bounds__(256) void final_lin(const float* __restrict__ pooled,
                                                 const float* __restrict__ lw,
                                                 const float* __restrict__ lb,
                                                 float* __restrict__ out) {
    int t = blockIdx.x * 256 + threadIdx.x;
    if (t >= NGRAPH * NOUT) return;
    int g = t / NOUT, o = t % NOUT;
    const float* p = pooled + (size_t)g * C;
    const float* w = lw + (size_t)o * C;
    float s = 0.f;
#pragma unroll 8
    for (int c = 0; c < C; c++) s = fmaf(p[c], w[c], s);
    out[t] = s + lb[o];
}

// ---------------------------------------------------------------------------
extern "C" void kernel_launch(void* const* d_in, const int* in_sizes, int n_in,
                              void* d_out, int out_size, void* d_ws, size_t ws_size,
                              hipStream_t stream) {
    const float* x     = (const float*)d_in[0];
    const int*   ei    = (const int*)  d_in[1];
    const float* ew    = (const float*)d_in[2];
    const int*   batch = (const int*)  d_in[3];
    const float* We[2]   = { (const float*)d_in[4],  (const float*)d_in[12] };
    const float* be[2]   = { (const float*)d_in[5],  (const float*)d_in[13] };
    const float* W1f[2]  = { (const float*)d_in[6],  (const float*)d_in[14] };
    const float* b1[2]   = { (const float*)d_in[7],  (const float*)d_in[15] };
    const float* gm[2]   = { (const float*)d_in[8],  (const float*)d_in[16] };
    const float* bt[2]   = { (const float*)d_in[9],  (const float*)d_in[17] };
    const float* W2f[2]  = { (const float*)d_in[10], (const float*)d_in[18] };
    const float* b2[2]   = { (const float*)d_in[11], (const float*)d_in[19] };
    const float* lin_w = (const float*)d_in[20];
    const float* lin_b = (const float*)d_in[21];

    char* wsb = (char*)d_ws;
    const size_t BIG = (size_t)NODES * C * sizeof(float);          // 25,600,000 B
    float*          B0   = (float*)wsb;              // aggr1 -> x1 -> hidden2
    float*          B1   = (float*)(wsb + BIG);      // hidden1 -> aggr2
    unsigned short* WBF  = (unsigned short*)(wsb + 2 * BIG);
    float*          BN   = (float*)(wsb + 2 * BIG + 131072);
    float*          POOL = (float*)((char*)BN + 2048);
    unsigned short* W1b[2] = { WBF,          WBF + 2 * 16384 };
    unsigned short* W2b[2] = { WBF + 16384,  WBF + 3 * 16384 };
    float* SC = BN + 256;   // bn scale
    float* SH = BN + 384;   // bn shift

    const int GEMM_BLKS = (MTILES * 32 + 255) / 256;   // 391

    cvt_w<<<256, 256, 0, stream>>>(W1f[0], W2f[0], W1f[1], W2f[1], WBF);
    hipMemsetAsync(POOL, 0, (size_t)NGRAPH * C * sizeof(float), stream);

    // ---------------- Layer 0 ----------------
    hipMemsetAsync(B0, 0, BIG, stream);                // aggr1 = 0
    hipMemsetAsync(BN, 0, 1024, stream);               // bn sum/sumsq = 0
    msg_scatter<<<EDGES / 8, 256, 0, stream>>>(x, ei, ew, We[0], be[0], B0);
    // hidden1 = bf16(x + aggr1) @ W1^T + b1  (+ BN stats)
    gemm_fused<1, 0><<<GEMM_BLKS, 256, 0, stream>>>(x, B0, nullptr, nullptr,
                                                    W1b[0], b1[0], B1,
                                                    BN, BN + 128, nullptr, nullptr);
    bn_fin<<<1, 128, 0, stream>>>(BN, BN + 128, gm[0], bt[0], SC, SH);
    // x1 = relu( bf16(relu(hidden1*sc+sh)) @ W2^T + b2 )  -> B0
    gemm_fused<2, 1><<<GEMM_BLKS, 256, 0, stream>>>(B1, nullptr, SC, SH,
                                                    W2b[0], b2[0], B0,
                                                    nullptr, nullptr, nullptr, nullptr);

    // ---------------- Layer 1 ----------------
    hipMemsetAsync(B1, 0, BIG, stream);                // aggr2 = 0
    hipMemsetAsync(BN, 0, 1024, stream);
    msg_scatter<<<EDGES / 8, 256, 0, stream>>>(B0, ei, ew, We[1], be[1], B1);
    // hidden2 = bf16(x1 + aggr2) @ W1^T + b1 -> B0 (per-wave same-row overwrite of x1;
    // stores depend on accumulators which depend on all loads -> ordered)
    gemm_fused<1, 0><<<GEMM_BLKS, 256, 0, stream>>>(B0, B1, nullptr, nullptr,
                                                    W1b[1], b1[1], B0,
                                                    BN, BN + 128, nullptr, nullptr);
    bn_fin<<<1, 128, 0, stream>>>(BN, BN + 128, gm[1], bt[1], SC, SH);
    // pooled = segment_max( relu( bf16(relu(hidden2*sc+sh)) @ W2^T + b2 ) )
    gemm_fused<2, 2><<<GEMM_BLKS, 256, 0, stream>>>(B0, nullptr, SC, SH,
                                                    W2b[1], b2[1], nullptr,
                                                    nullptr, nullptr, batch, POOL);

    final_lin<<<(NGRAPH * NOUT + 255) / 256, 256, 0, stream>>>(POOL, lin_w, lin_b,
                                                               (float*)d_out);
}
